// SE3MixAttention_17136919511779
// MI455X (gfx1250) — matrix operand order repaired and checked by reading, not compiled, not run on hardware
//
#include <hip/hip_runtime.h>

typedef __attribute__((ext_vector_type(16))) __bf16 v16bf;
typedef __attribute__((ext_vector_type(8)))  float  v8f;

#define DEV static __device__ __forceinline__

namespace {
constexpr int kB   = 128;   // batches
constexpr int kNN  = 64;    // nodes per batch
constexpr int kD   = 64;    // feature dim
constexpr int kD2  = 128;   // 2*D
constexpr int kN   = kB * kNN;       // 8192 nodes
constexpr int kDT  = 8;              // dst-tile size
constexpr int kEPB = kNN * kDT;      // 512 edge rows per block

// workspace offsets (in floats)
constexpr size_t oXn = 0;
constexpr size_t oQn = oXn + (size_t)kN * 3;
constexpr size_t oKV = oQn + (size_t)kN * kD;
constexpr size_t oA  = oKV + (size_t)kN * kD2;
} // namespace

// Fast hardware approximations (v_rcp_f32 / v_rsq_f32 / v_sqrt_f32, ~1 ulp)
DEV float frcp(float x)  { return __builtin_amdgcn_rcpf(x); }
DEV float frsq(float x)  { return __builtin_amdgcn_rsqf(x); }
DEV float fsqrt(float x) { return __builtin_amdgcn_sqrtf(x); }

DEV float silu_f(float x) { return x * frcp(1.0f + __expf(-x)); }

DEV v8f wmma_bf16(v16bf a, v16bf b, v8f c) {
  // v_wmma_f32_16x16x32_bf16 (codegen-confirmed signature, 8 args)
  return __builtin_amdgcn_wmma_f32_16x16x32_bf16(false, a, false, b, (short)0, c,
                                                 false, false);
}

// A fragment (16x32 bf16, rows m0..m0+15, K-chunk kk) from row-major LDS, stride ld.
// ISA 7.12.2: lanes0-15 row M=lane, elems0-7=K0..7, elems8-15=K16..23;
//             lanes16-31 row M=lane-16, elems0-7=K8..15, elems8-15=K24..31.
DEV v16bf ldsA(const __bf16* A, int ld, int m0, int kk) {
  int lane = threadIdx.x & 31;
  int hb = lane >> 4;
  int m = m0 + (lane & 15);
  v16bf a;
#pragma unroll
  for (int e = 0; e < 16; ++e) {
    int k = kk + ((e < 8) ? e : e + 8) + (hb ? 8 : 0);
    a[e] = A[m * ld + k];
  }
  return a;
}

// B fragment (32x16 bf16, K-chunk kk, cols n0..n0+15) from row-major [K][N] LDS.
// lanes0-15: col N=lane, elems0-7=K0..7, elems8-15=K16..23; lanes16-31: col N=lane-16, elems0-7=K8..15, elems8-15=K24..31.
DEV v16bf ldsB(const __bf16* Bm, int ld, int kk, int n0) {
  int lane = threadIdx.x & 31;
  int hb = lane >> 4;
  int n = n0 + (lane & 15);
  v16bf b;
#pragma unroll
  for (int e = 0; e < 16; ++e) {
    int k = kk + ((e < 8) ? e : e + 8) + (hb ? 8 : 0);
    b[e] = Bm[k * ld + n];
  }
  return b;
}

DEV float wred_add(float v) {
#pragma unroll
  for (int o = 16; o; o >>= 1) v += __shfl_xor(v, o, 32);
  return v;
}
DEV float wred_max(float v) {
#pragma unroll
  for (int o = 16; o; o >>= 1) v = fmaxf(v, __shfl_xor(v, o, 32));
  return v;
}

// ---------------------------------------------------------------------------
// Kernel 1: per-node precompute. 1 block per batch, 256 threads (8 waves).
//   mean_norm, Xn; Qn = LN(silu(H@qw1)@qw2); KVbase = H @ kvw1[1:,:]
// ---------------------------------------------------------------------------
__global__ void node_pre_kernel(const float* __restrict__ X,
                                const float* __restrict__ H,
                                const float* __restrict__ xw,
                                const float* __restrict__ qw1,
                                const float* __restrict__ qw2,
                                const float* __restrict__ kvw1,
                                const float* __restrict__ qnw,
                                const float* __restrict__ qnb,
                                float* __restrict__ Xn,
                                float* __restrict__ Qn,
                                float* __restrict__ KV) {
  extern __shared__ char smem[];
  float*  Qbuf = (float*)smem;            // 64*64
  float*  red  = Qbuf + 64 * 64;          // 64
  __bf16* Hs   = (__bf16*)(red + 64);     // 64*64
  __bf16* W1   = Hs + 64 * 64;            // 64*64
  __bf16* W2   = W1 + 64 * 64;            // 64*64
  __bf16* Wkv  = W2 + 64 * 64;            // 64*128
  __bf16* T1s  = Wkv + 64 * 128;          // 64*64

  const int tid = threadIdx.x;
  const int b = blockIdx.x;
  const int nb = b * kNN;
  const int w = tid >> 5, lane = tid & 31, hb = lane >> 4, ln = lane & 15;

  for (int i = tid; i < 64 * 64; i += 256) {
    Hs[i] = (__bf16)H[(size_t)nb * kD + i];
    W1[i] = (__bf16)qw1[i];
    W2[i] = (__bf16)qw2[i];
  }
  for (int i = tid; i < 64 * 128; i += 256) Wkv[i] = (__bf16)kvw1[kD2 + i];
  if (tid < 64) {
    float x0 = X[(nb + tid) * 3 + 0];
    float x1 = X[(nb + tid) * 3 + 1];
    float x2 = X[(nb + tid) * 3 + 2];
    red[tid] = fsqrt(x0 * x0 + x1 * x1 + x2 * x2);
  }
  __syncthreads();
  if (tid == 0) {
    float s = 0.f;
    for (int i = 0; i < 64; ++i) s += red[i];
    red[0] = xw[0] * frcp(s * (1.0f / 64.0f) + 1e-5f);  // combined scale
  }
  __syncthreads();
  if (tid < 64) {
    float sc = red[0];
#pragma unroll
    for (int c = 0; c < 3; ++c)
      Xn[(size_t)(nb + tid) * 3 + c] = X[(size_t)(nb + tid) * 3 + c] * sc;
  }
  __syncthreads();

  // T1 = silu(H @ qw1)
  for (int t = w; t < 16; t += 8) {
    int m0 = (t >> 2) * 16, n0 = (t & 3) * 16;
    v8f c = {};
    for (int kk = 0; kk < 64; kk += 32)
      c = wmma_bf16(ldsA(Hs, 64, m0, kk), ldsB(W1, 64, kk, n0), c);
#pragma unroll
    for (int r = 0; r < 8; ++r)
      T1s[(m0 + r + 8 * hb) * 64 + n0 + ln] = (__bf16)silu_f(c[r]);
  }
  __syncthreads();

  // Q = T1 @ qw2 -> Qbuf (f32)
  for (int t = w; t < 16; t += 8) {
    int m0 = (t >> 2) * 16, n0 = (t & 3) * 16;
    v8f c = {};
    for (int kk = 0; kk < 64; kk += 32)
      c = wmma_bf16(ldsA(T1s, 64, m0, kk), ldsB(W2, 64, kk, n0), c);
#pragma unroll
    for (int r = 0; r < 8; ++r) Qbuf[(m0 + r + 8 * hb) * 64 + n0 + ln] = c[r];
  }
  __syncthreads();

  // LayerNorm rows of Q -> Qn (global). 8 rows per wave.
  for (int rr = 0; rr < 8; ++rr) {
    int row = w * 8 + rr;
    float v0 = Qbuf[row * 64 + lane];
    float v1 = Qbuf[row * 64 + lane + 32];
    float s = wred_add(v0 + v1);
    float s2 = wred_add(v0 * v0 + v1 * v1);
    float mu = s * (1.0f / 64.0f);
    float var = s2 * (1.0f / 64.0f) - mu * mu;
    float rstd = frsq(var + 1e-5f);
    Qn[(size_t)(nb + row) * kD + lane] = (v0 - mu) * rstd * qnw[lane] + qnb[lane];
    Qn[(size_t)(nb + row) * kD + lane + 32] =
        (v1 - mu) * rstd * qnw[lane + 32] + qnb[lane + 32];
  }

  // KVbase = H @ kvw1[1:,:]  (64x128)
  for (int t = w; t < 32; t += 8) {
    int m0 = (t >> 3) * 16, n0 = (t & 7) * 16;
    v8f c = {};
    for (int kk = 0; kk < 64; kk += 32)
      c = wmma_bf16(ldsA(Hs, 64, m0, kk), ldsB(Wkv, 128, kk, n0), c);
#pragma unroll
    for (int r = 0; r < 8; ++r)
      KV[(size_t)(nb + m0 + r + 8 * hb) * kD2 + n0 + ln] = c[r];
  }
}

// ---------------------------------------------------------------------------
// Kernel 2: fused edge attention. 1 block per (batch, 8-dst tile): 1024 blocks.
// Edge row r = s*8 + dd  (s in 0..63, dd in 0..7), 512 rows.
// ---------------------------------------------------------------------------
__global__ void batch_att_kernel(const float* __restrict__ Xn,
                                 const float* __restrict__ Qn,
                                 const float* __restrict__ KV,
                                 const float* __restrict__ kvw1,
                                 const float* __restrict__ kvw2,
                                 const float* __restrict__ knw,
                                 const float* __restrict__ knb,
                                 const float* __restrict__ phxw1,
                                 const float* __restrict__ phxw2,
                                 float* __restrict__ Aout,
                                 float* __restrict__ Xout) {
  extern __shared__ char smem[];
  float* KVs  = (float*)smem;        // 64*128
  float* Qd   = KVs + 64 * 128;      // 8*64
  float* Xns  = Qd + 8 * 64;         // 64*3
  float* reld = Xns + 64 * 3;        // 512
  float* xrel = reld + 512;          // 512*3
  float* scr  = xrel + 512 * 3;      // 8 waves * 16 * 64
  float* sa   = scr + 8 * 16 * 64;   // 512 scores -> alphas
  float* phxb = sa + 512;            // 512
  float* kvr0 = phxb + 512;          // 128
  float* pw2  = kvr0 + 128;          // 64
  float* knwv = pw2 + 64;            // 64
  float* knbv = knwv + 64;           // 64
  __bf16* Wkv2 = (__bf16*)(knbv + 64);  // 128*128
  __bf16* Wphx = Wkv2 + 128 * 128;      // 64*64
  __bf16* Vb   = Wphx + 64 * 64;        // 512*64

  const int tid = threadIdx.x;
  const int b = blockIdx.x >> 3;
  const int dbase = (blockIdx.x & 7) * kDT;
  const int nb = b * kNN;
  const int w = tid >> 5, lane = tid & 31, hb = lane >> 4, ln = lane & 15;

  for (int i = tid; i < 64 * 128; i += 256) KVs[i] = KV[(size_t)nb * kD2 + i];
  for (int i = tid; i < 8 * 64; i += 256) Qd[i] = Qn[(size_t)(nb + dbase) * kD + i];
  for (int i = tid; i < 64 * 3; i += 256) Xns[i] = Xn[(size_t)nb * 3 + i];
  for (int i = tid; i < 128 * 128; i += 256) Wkv2[i] = (__bf16)kvw2[i];
  for (int i = tid; i < 64 * 64; i += 256) Wphx[i] = (__bf16)phxw1[i];
  if (tid < 128) kvr0[tid] = kvw1[tid];
  if (tid < 64) { pw2[tid] = phxw2[tid]; knwv[tid] = knw[tid]; knbv[tid] = knb[tid]; }
  __syncthreads();

  // Edge geometry: rel_dist, X_rel_norm
  for (int e = tid; e < kEPB; e += 256) {
    int s = e >> 3, d = dbase + (e & 7);
    float r0 = Xns[s * 3 + 0] - Xns[d * 3 + 0];
    float r1 = Xns[s * 3 + 1] - Xns[d * 3 + 1];
    float r2 = Xns[s * 3 + 2] - Xns[d * 3 + 2];
    float rd = r0 * r0 + r1 * r1 + r2 * r2;
    reld[e] = rd;
    float inv = frcp(1.0f + fsqrt(rd + 1e-8f));
    xrel[e * 3 + 0] = r0 * inv;
    xrel[e * 3 + 1] = r1 * inv;
    xrel[e * 3 + 2] = r2 * inv;
  }
  __syncthreads();

  // Pass 1: kv = silu(kv_pre) @ kvw2 ; LN(K) + scores ; stash V (bf16)
  for (int et = w; et < 32; et += 8) {
    const int r0row = et * 16;
    v16bf afr[4];
#pragma unroll
    for (int kc = 0; kc < 4; ++kc) {
      int kk = kc * 32;
      int r = r0row + ln;
      int s = r >> 3;
      float rd = reld[r];
      v16bf a;
#pragma unroll
      for (int e = 0; e < 16; ++e) {
        int j = kk + ((e < 8) ? e : e + 8) + (hb ? 8 : 0);
        a[e] = (__bf16)silu_f(rd * kvr0[j] + KVs[s * kD2 + j]);
      }
      afr[kc] = a;
    }
    float* S = scr + w * 16 * 64;
#pragma unroll
    for (int ct = 0; ct < 8; ++ct) {
      v8f c = {};
#pragma unroll
      for (int kc = 0; kc < 4; ++kc)
        c = wmma_bf16(afr[kc], ldsB(Wkv2, 128, kc * 32, ct * 16), c);
      if (ct < 4) {
#pragma unroll
        for (int r = 0; r < 8; ++r) S[(r + 8 * hb) * 64 + ct * 16 + ln] = c[r];
      } else {
#pragma unroll
        for (int r = 0; r < 8; ++r)
          Vb[(r0row + r + 8 * hb) * 64 + (ct - 4) * 16 + ln] = (__bf16)c[r];
      }
    }
    // LayerNorm(K) + score (dot with Q[dst]) for 16 rows; lane pair (m, m+16)
    {
      int m = ln;                 // row within tile
      int r = r0row + m;
      int dd = r & 7;
      float sum = 0.f, sq = 0.f;
#pragma unroll
      for (int j0 = 0; j0 < 32; ++j0) {
        float v = S[m * 64 + hb * 32 + j0];
        sum += v; sq += v * v;
      }
      sum += __shfl_xor(sum, 16, 32);
      sq += __shfl_xor(sq, 16, 32);
      float mu = sum * (1.0f / 64.0f);
      float var = sq * (1.0f / 64.0f) - mu * mu;
      float rstd = frsq(var + 1e-5f);
      float dotp = 0.f;
#pragma unroll
      for (int j0 = 0; j0 < 32; ++j0) {
        int j = hb * 32 + j0;
        float kh = (S[m * 64 + j] - mu) * rstd * knwv[j] + knbv[j];
        dotp += kh * Qd[dd * 64 + j];
      }
      dotp += __shfl_xor(dotp, 16, 32);
      if (hb == 0) sa[r] = dotp * 0.125f;  // / sqrt(64)
    }
  }
  __syncthreads();

  // Softmax over src (exclude s==d). wave w handles dst dd=w.
  {
    int dd = w, d = dbase + dd;
    int s0 = lane, s1 = lane + 32;
    float v0 = (s0 == d) ? -1e30f : sa[s0 * 8 + dd];
    float v1 = (s1 == d) ? -1e30f : sa[s1 * 8 + dd];
    float mx = wred_max(fmaxf(v0, v1));
    float e0 = (s0 == d) ? 0.f : __expf(v0 - mx);
    float e1 = (s1 == d) ? 0.f : __expf(v1 - mx);
    float inv = frcp(wred_add(e0 + e1));
    sa[s0 * 8 + dd] = e0 * inv;
    sa[s1 * 8 + dd] = e1 * inv;
  }
  __syncthreads();

  // A[d] = sum_s alpha * V   (wave w -> dst dd=w)
  {
    int dd = w;
    float a0 = 0.f, a1 = 0.f;
    for (int s = 0; s < 64; ++s) {
      float al = sa[s * 8 + dd];
      int rb = (s * 8 + dd) * 64;
      a0 += al * (float)Vb[rb + lane];
      a1 += al * (float)Vb[rb + lane + 32];
    }
    size_t node = (size_t)(nb + dbase + dd);
    Aout[node * kD + lane] = a0;
    Aout[node * kD + lane + 32] = a1;
  }

  // Pass 2: phx = silu(att @ phxw1) . phxw2 ; att = alpha * V (built in regs)
  for (int et = w; et < 32; et += 8) {
    const int r0row = et * 16;
    v16bf afr[2];
#pragma unroll
    for (int kc = 0; kc < 2; ++kc) {
      int kk = kc * 32;
      int r = r0row + ln;
      float al = sa[r];
      v16bf a;
#pragma unroll
      for (int e = 0; e < 16; ++e) {
        int j = kk + ((e < 8) ? e : e + 8) + (hb ? 8 : 0);
        a[e] = (__bf16)(al * (float)Vb[r * 64 + j]);
      }
      afr[kc] = a;
    }
    float* S = scr + w * 16 * 64;
#pragma unroll
    for (int ct = 0; ct < 4; ++ct) {
      v8f c = {};
#pragma unroll
      for (int kc = 0; kc < 2; ++kc)
        c = wmma_bf16(afr[kc], ldsB(Wphx, 64, kc * 32, ct * 16), c);
#pragma unroll
      for (int r = 0; r < 8; ++r) S[(r + 8 * hb) * 64 + ct * 16 + ln] = c[r];
    }
    {
      int m = ln;
      float acc = 0.f;
#pragma unroll
      for (int j0 = 0; j0 < 32; ++j0) {
        int j = hb * 32 + j0;
        acc += silu_f(S[m * 64 + j]) * pw2[j];
      }
      acc += __shfl_xor(acc, 16, 32);
      if (hb == 0) phxb[r0row + m] = acc;
    }
  }
  __syncthreads();

  // X_out[d] = Xn[d] + sum_s xrel * phx    (wave w -> dst dd=w)
  {
    int dd = w, d = dbase + dd;
    float x0 = 0.f, x1 = 0.f, x2 = 0.f;
    for (int s = lane; s < 64; s += 32) {
      int e = s * 8 + dd;
      float p = phxb[e];
      x0 += xrel[e * 3 + 0] * p;
      x1 += xrel[e * 3 + 1] * p;
      x2 += xrel[e * 3 + 2] * p;
    }
    x0 = wred_add(x0); x1 = wred_add(x1); x2 = wred_add(x2);
    if (lane == 0) {
      size_t node = (size_t)(nb + d);
      Xout[node * 3 + 0] = Xns[d * 3 + 0] + x0;
      Xout[node * 3 + 1] = Xns[d * 3 + 1] + x1;
      Xout[node * 3 + 2] = Xns[d * 3 + 2] + x2;
    }
  }
}

// ---------------------------------------------------------------------------
// Kernel 3: H_out = H + silu((A*A*H) @ phhw1 + b1) @ phhw2 + b2   (per node)
// (uses AH = A ⊙ H[dst] identity). 1 block per batch.
// ---------------------------------------------------------------------------
__global__ void hout_kernel(const float* __restrict__ H,
                            const float* __restrict__ Aws,
                            const float* __restrict__ w1,
                            const float* __restrict__ b1,
                            const float* __restrict__ w2,
                            const float* __restrict__ b2,
                            float* __restrict__ Hout) {
  extern __shared__ char smem[];
  float*  Hs  = (float*)smem;        // 64*64
  float*  b1v = Hs + 64 * 64;        // 64
  float*  b2v = b1v + 64;            // 64
  __bf16* U   = (__bf16*)(b2v + 64); // 64*64
  __bf16* W1b = U + 64 * 64;
  __bf16* W2b = W1b + 64 * 64;
  __bf16* Ts  = W2b + 64 * 64;

  const int tid = threadIdx.x;
  const int nb = blockIdx.x * kNN;
  const int w = tid >> 5, lane = tid & 31, hb = lane >> 4, ln = lane & 15;

  for (int i = tid; i < 64 * 64; i += 256) {
    float h = H[(size_t)nb * kD + i];
    float a = Aws[(size_t)nb * kD + i];
    Hs[i] = h;
    U[i] = (__bf16)(a * a * h);
    W1b[i] = (__bf16)w1[i];
    W2b[i] = (__bf16)w2[i];
  }
  if (tid < 64) { b1v[tid] = b1[tid]; b2v[tid] = b2[tid]; }
  __syncthreads();

  for (int t = w; t < 16; t += 8) {
    int m0 = (t >> 2) * 16, n0 = (t & 3) * 16;
    v8f c = {};
    for (int kk = 0; kk < 64; kk += 32)
      c = wmma_bf16(ldsA(U, 64, m0, kk), ldsB(W1b, 64, kk, n0), c);
#pragma unroll
    for (int r = 0; r < 8; ++r)
      Ts[(m0 + r + 8 * hb) * 64 + n0 + ln] = (__bf16)silu_f(c[r] + b1v[n0 + ln]);
  }
  __syncthreads();

  for (int t = w; t < 16; t += 8) {
    int m0 = (t >> 2) * 16, n0 = (t & 3) * 16;
    v8f c = {};
    for (int kk = 0; kk < 64; kk += 32)
      c = wmma_bf16(ldsA(Ts, 64, m0, kk), ldsB(W2b, 64, kk, n0), c);
#pragma unroll
    for (int r = 0; r < 8; ++r) {
      int row = m0 + r + 8 * hb, col = n0 + ln;
      Hout[(size_t)(nb + row) * kD + col] = Hs[row * 64 + col] + c[r] + b2v[col];
    }
  }
}

// ---------------------------------------------------------------------------
extern "C" void kernel_launch(void* const* d_in, const int* in_sizes, int n_in,
                              void* d_out, int out_size, void* d_ws, size_t ws_size,
                              hipStream_t stream) {
  (void)in_sizes; (void)n_in; (void)out_size; (void)ws_size;
  const float* X     = (const float*)d_in[1];
  const float* H     = (const float*)d_in[2];
  const float* xw    = (const float*)d_in[4];
  const float* qw1   = (const float*)d_in[5];
  const float* qw2   = (const float*)d_in[6];
  const float* kvw1  = (const float*)d_in[7];
  const float* kvw2  = (const float*)d_in[8];
  const float* qnw   = (const float*)d_in[9];
  const float* qnb   = (const float*)d_in[10];
  const float* knw   = (const float*)d_in[11];
  const float* knb   = (const float*)d_in[12];
  const float* phxw1 = (const float*)d_in[13];
  const float* phxw2 = (const float*)d_in[14];
  const float* phhw1 = (const float*)d_in[15];
  const float* phhb1 = (const float*)d_in[16];
  const float* phhw2 = (const float*)d_in[17];
  const float* phhb2 = (const float*)d_in[18];

  float* ws  = (float*)d_ws;
  float* Xn  = ws + oXn;
  float* Qn  = ws + oQn;
  float* KV  = ws + oKV;
  float* A   = ws + oA;
  float* out = (float*)d_out;
  float* Xout = out;                    // N*3
  float* Hout = out + (size_t)kN * 3;   // N*D

  constexpr size_t smem1 = (64 * 64 + 64) * 4 + (3 * 64 * 64 + 64 * 128 + 64 * 64) * 2;
  constexpr size_t smem2 =
      (64 * 128 + 8 * 64 + 64 * 3 + 512 + 512 * 3 + 8 * 16 * 64 + 512 + 512 + 128 +
       3 * 64) * 4 +
      (128 * 128 + 64 * 64 + 512 * 64) * 2;
  constexpr size_t smem3 = (64 * 64 + 128) * 4 + (4 * 64 * 64) * 2;

  node_pre_kernel<<<kB, 256, smem1, stream>>>(X, H, xw, qw1, qw2, kvw1, qnw, qnb,
                                              Xn, Qn, KV);
  batch_att_kernel<<<kB * (kNN / kDT), 256, smem2, stream>>>(
      Xn, Qn, KV, kvw1, kvw2, knw, knb, phxw1, phxw2, A, Xout);
  hout_kernel<<<kB, 256, smem3, stream>>>(H, A, phhw1, phhb1, phhw2, phhb2, Hout);
}
